// MambNet1_87067577025385
// MI455X (gfx1250) — matrix operand order repaired and checked by reading, not compiled, not run on hardware
//
#include <hip/hip_runtime.h>
#include <hip/hip_bf16.h>
#include <math.h>

// ---------------- types / helpers ----------------
typedef __attribute__((ext_vector_type(16))) __bf16 v16bf;
typedef __attribute__((ext_vector_type(8)))  float  v8f;
typedef __attribute__((ext_vector_type(16))) unsigned short v16us;
typedef __attribute__((ext_vector_type(4)))  int    v4i;
typedef __attribute__((address_space(1))) v4i* gpv4;   // global-AS int4 ptr
typedef __attribute__((address_space(3))) v4i* spv4;   // LDS-AS int4 ptr

#define DEV __device__ __forceinline__
#define CDIV(a,b) (((a)+(b)-1)/(b))

DEV unsigned short f2b(float f){ union{float f;unsigned u;}x; x.f=f; unsigned r=x.u + 0x7FFFu + ((x.u>>16)&1u); return (unsigned short)(r>>16); }
DEV float b2f(unsigned short h){ union{unsigned u;float f;}x; x.u=((unsigned)h)<<16; return x.f; }

union FragU { v16us v; unsigned short u[16]; };
DEV v16bf bfv(FragU f){ union{v16us u; v16bf b;}c; c.u=f.v; return c.b; }
DEV v8f wmma_bf(FragU a, FragU b, v8f c){
  return __builtin_amdgcn_wmma_f32_16x16x32_bf16(false, bfv(a), false, bfv(b), (short)0, c, false, false);
}

// --------- CDNA5 async global->LDS copy (16B per lane, ASYNCcnt-tracked) ---------
DEV void async_cp16(void* lds, const void* g){
#if __has_builtin(__builtin_amdgcn_global_load_async_to_lds_b128)
  __builtin_amdgcn_global_load_async_to_lds_b128((gpv4)g, (spv4)lds, 0, 0);
#else
  unsigned lo = (unsigned)(size_t)lds;
  asm volatile("global_load_async_to_lds_b128 %0, %1, off"
               :: "v"(lo), "v"((unsigned long long)(size_t)g)
               : "memory");
#endif
}
DEV void async_wait(){
#if __has_builtin(__builtin_amdgcn_s_wait_asynccnt)
  __builtin_amdgcn_s_wait_asynccnt(0);
#else
  asm volatile("s_wait_asynccnt 0" ::: "memory");
#endif
}

// ---------------- small utility kernels ----------------
__global__ void k_cvt(const float* __restrict__ s, unsigned short* __restrict__ d, int n){
  int i = blockIdx.x*256 + threadIdx.x; if (i < n) d[i] = f2b(s[i]);
}
__global__ void k_zero16(unsigned short* __restrict__ d, long long n){
  long long i = (long long)blockIdx.x*256 + threadIdx.x; if (i < n) d[i] = 0;
}
__global__ void k_rev(const unsigned short* __restrict__ s, unsigned short* __restrict__ d, int lp, int total){
  int i = blockIdx.x*256 + threadIdx.x; if (i >= total) return;
  int c = i & 63; int t = (i >> 6) % lp; int b = i / (64*lp);
  d[((size_t)b*lp + t)*64 + c] = s[((size_t)b*lp + (lp-1-t))*64 + c];
}
__global__ void k_revcat(const unsigned short* __restrict__ s, unsigned short* __restrict__ d, int lp, int total){
  int i = blockIdx.x*256 + threadIdx.x; if (i >= total) return;
  int c = i & 63; int t = (i >> 6) % lp; int b = i / (64*lp);
  d[((size_t)b*lp + t)*128 + 64 + c] = s[((size_t)b*lp + (lp-1-t))*64 + c];
}

// ---------------- generic WMMA GEMM: C[M,N] = A[M,K] * W[N,K]^T ----------------
// Requirements on async path: M%64==0, K%32==0, W buffer padded to NT-multiple rows.
// EPI: 0 f32 plain | 1 bf16 plain | 2 bf16 relu(acc*p0[n]+p1[n]) | 3 bf16 acc+p0[n]
template<int EPI, int NT>
__global__ __launch_bounds__(256) void k_gemm(
    const unsigned short* __restrict__ A, const unsigned short* __restrict__ W,
    void* __restrict__ Cv, int M, int N, int K, int lda, int ldw, int ldc,
    const float* __restrict__ p0, const float* __restrict__ p1)
{
  constexpr int TPW = NT/32;           // C tiles per wave
  __shared__ unsigned short As[64][32];
  __shared__ unsigned short Ws[NT][32];
  const int tid = threadIdx.x, lane = tid & 31, wv = tid >> 5;
  const int bm = blockIdx.y * 64, bn = blockIdx.x * NT;
  const int tm = wv & 3, tnb = (wv >> 2) * TPW;
  const int half = lane >> 4, l16 = lane & 15;
  const int r = tid >> 2, cc = (tid & 3) * 8;
  v8f acc[TPW];
  { v8f z = {}; for (int t=0;t<TPW;++t) acc[t]=z; }
  for (int k0 = 0; k0 < K; k0 += 32) {
    async_cp16(&As[r][cc], &A[(size_t)(bm+r)*lda + k0 + cc]);
    for (int q=0; q<NT/64; ++q)
      async_cp16(&Ws[r+q*64][cc], &W[(size_t)(bn+r+q*64)*ldw + k0 + cc]);
    async_wait();
    __syncthreads();
    FragU a;
    int ar = tm*16 + l16;
    for (int i=0;i<8;++i){ a.u[i] = As[ar][half*16+i]; a.u[8+i] = As[ar][half*16+8+i]; }
    for (int t=0;t<TPW;++t){
      FragU b;
      int c0 = (tnb+t)*16 + l16;
      for (int i=0;i<16;++i) b.u[i] = Ws[c0][half*16+i];
      acc[t] = wmma_bf(a, b, acc[t]);
    }
    __syncthreads();
  }
  for (int t=0;t<TPW;++t){
    int nn = bn + (tnb+t)*16 + l16;
    for (int j=0;j<8;++j){
      int mm = bm + tm*16 + j + 8*half;
      if (mm < M && nn < N) {
        float v = acc[t][j];
        size_t off = (size_t)mm*ldc + nn;
        if (EPI==0) ((float*)Cv)[off] = v;
        else if (EPI==1) ((unsigned short*)Cv)[off] = f2b(v);
        else if (EPI==2) { v = v*p0[nn] + p1[nn]; ((unsigned short*)Cv)[off] = f2b(v>0.f?v:0.f); }
        else ((unsigned short*)Cv)[off] = f2b(v + p0[nn]);
      }
    }
  }
}

// ---------------- implicit-im2col conv3x3 + BN-affine + ReLU (64x128 block) ----------------
// variant A: two NCHW f32 sources concatenated on channels
__global__ __launch_bounds__(256) void k_conv_nchw2(
    const float* __restrict__ Ain, const float* __restrict__ Bin,
    const unsigned short* __restrict__ W, unsigned short* __restrict__ Out,
    const float* __restrict__ gamma, const float* __restrict__ beta,
    int cin, int s, int lp)
{
  const int l = s*s, M = 16*l, K = 2*cin*9;
  __shared__ unsigned short As[64][32];
  __shared__ unsigned short Ws[128][32];
  const int tid=threadIdx.x, lane=tid&31, wv=tid>>5;
  const int bm=blockIdx.y*64, bn=blockIdx.x*128;
  const int tm=wv&3, tnb=(wv>>2)*4, half=lane>>4, l16=lane&15;
  const int r = tid>>2, cc=(tid&3)*8;
  const int m = bm + r;
  int b=0,y=0,x=0;
  if (m < M){ b=m/l; int pos=m-b*l; y=pos/s; x=pos-y*s; }
  v8f acc[4]; { v8f z={}; for(int t=0;t<4;++t) acc[t]=z; }
  for (int k0=0;k0<K;k0+=32){
    for (int q=0;q<2;++q)
      async_cp16(&Ws[r+q*64][cc], &W[(size_t)(bn+r+q*64)*K + k0 + cc]);
    for (int i=0;i<8;++i){
      int kk=k0+cc+i;
      float v=0.f;
      if (m<M && kk<K){
        int ci=kk/9, rem=kk-ci*9, kh=rem/3, kw=rem-kh*3;
        int iy=y+kh-1, ix=x+kw-1;
        if (iy>=0&&iy<s&&ix>=0&&ix<s)
          v = (ci<cin) ? Ain[(((size_t)b*cin+ci)*s+iy)*s+ix]
                       : Bin[(((size_t)b*cin+(ci-cin))*s+iy)*s+ix];
      }
      As[r][cc+i]=f2b(v);
    }
    async_wait();
    __syncthreads();
    FragU a;
    int ar=tm*16+l16;
    for(int i=0;i<8;++i){ a.u[i]=As[ar][half*16+i]; a.u[8+i]=As[ar][half*16+8+i]; }
    for (int t=0;t<4;++t){
      FragU bb;
      int c0=(tnb+t)*16+l16;
      for(int i=0;i<16;++i) bb.u[i]=Ws[c0][half*16+i];
      acc[t]=wmma_bf(a,bb,acc[t]);
    }
    __syncthreads();
  }
  for(int t=0;t<4;++t){
    int nn = bn + (tnb+t)*16 + l16;
    for(int j=0;j<8;++j){
      int mm = bm + tm*16 + j + 8*half;
      if (mm<M && nn<128){
        int bb=mm/l, pos=mm-bb*l;
        float v = acc[t][j]*gamma[nn] + beta[nn];
        Out[((size_t)bb*lp+pos)*128 + nn] = f2b(v>0.f?v:0.f);
      }
    }
  }
}

// variant B: channels-last bf16 source [16*l, cin]
__global__ __launch_bounds__(256) void k_conv_cl(
    const unsigned short* __restrict__ X,
    const unsigned short* __restrict__ W, unsigned short* __restrict__ Out,
    const float* __restrict__ gamma, const float* __restrict__ beta,
    int cin, int s)
{
  const int l = s*s, M = 16*l, K = cin*9;
  __shared__ unsigned short As[64][32];
  __shared__ unsigned short Ws[128][32];
  const int tid=threadIdx.x, lane=tid&31, wv=tid>>5;
  const int bm=blockIdx.y*64, bn=blockIdx.x*128;
  const int tm=wv&3, tnb=(wv>>2)*4, half=lane>>4, l16=lane&15;
  const int r = tid>>2, cc=(tid&3)*8;
  const int m = bm + r;
  int b=0,y=0,x=0;
  if (m < M){ b=m/l; int pos=m-b*l; y=pos/s; x=pos-y*s; }
  v8f acc[4]; { v8f z={}; for(int t=0;t<4;++t) acc[t]=z; }
  for (int k0=0;k0<K;k0+=32){
    for (int q=0;q<2;++q)
      async_cp16(&Ws[r+q*64][cc], &W[(size_t)(bn+r+q*64)*K + k0 + cc]);
    for (int i=0;i<8;++i){
      int kk=k0+cc+i;
      unsigned short v=0;
      if (m<M && kk<K){
        int ci=kk/9, rem=kk-ci*9, kh=rem/3, kw=rem-kh*3;
        int iy=y+kh-1, ix=x+kw-1;
        if (iy>=0&&iy<s&&ix>=0&&ix<s)
          v = X[((size_t)b*l + iy*s + ix)*cin + ci];
      }
      As[r][cc+i]=v;
    }
    async_wait();
    __syncthreads();
    FragU a;
    int ar=tm*16+l16;
    for(int i=0;i<8;++i){ a.u[i]=As[ar][half*16+i]; a.u[8+i]=As[ar][half*16+8+i]; }
    for (int t=0;t<4;++t){
      FragU bb;
      int c0=(tnb+t)*16+l16;
      for(int i=0;i<16;++i) bb.u[i]=Ws[c0][half*16+i];
      acc[t]=wmma_bf(a,bb,acc[t]);
    }
    __syncthreads();
  }
  for(int t=0;t<4;++t){
    int nn = bn + (tnb+t)*16 + l16;
    for(int j=0;j<8;++j){
      int mm = bm + tm*16 + j + 8*half;
      if (mm<M && nn<128){
        float v = acc[t][j]*gamma[nn] + beta[nn];
        Out[(size_t)mm*128 + nn] = f2b(v>0.f?v:0.f);
      }
    }
  }
}

// ---------------- mamba elementwise ----------------
__global__ void k_dwconv(const unsigned short* __restrict__ proj,
    const float* __restrict__ cw, const float* __restrict__ cb,
    unsigned short* __restrict__ xconv, int lp, int total)
{
  int i = blockIdx.x*256 + threadIdx.x; if (i >= total) return;
  int chn = i % 384;
  int t = (i/384) % lp;
  int b = i/(384*lp);
  size_t rb = (size_t)b*lp;
  float acc = cb[chn];
  for (int j=0;j<4;++j){
    int tt = t - 3 + j;
    if (tt>=0) acc += cw[chn*4+j] * b2f(proj[(rb+tt)*528 + 128 + chn]);
  }
  float sv = acc / (1.f + expf(-acc));
  xconv[(rb+t)*384 + chn] = f2b(sv);
}

__global__ void k_dt(const unsigned short* __restrict__ proj,
    const float* __restrict__ dt_bias, const float* __restrict__ A_log,
    float* __restrict__ dtsp, float* __restrict__ dtA, int total)
{
  int i = blockIdx.x*256 + threadIdx.x; if (i >= total) return;
  int h = i & 1; size_t r = (size_t)(i >> 1);
  float v = b2f(proj[r*528 + 512 + h]) + dt_bias[h];
  float sp = v > 20.f ? v : log1pf(expf(v));
  dtsp[i] = sp;
  dtA[i] = -expf(A_log[h]) * sp;
}

// ---------------- SSD chunk kernel (intra-chunk, WMMA) ----------------
__global__ __launch_bounds__(256) void k_ssd_chunk(
    const unsigned short* __restrict__ xconv, const float* __restrict__ dtsp,
    const float* __restrict__ dtA, const float* __restrict__ Dh,
    float* __restrict__ Y, float* __restrict__ Sbuf,
    float* __restrict__ Asum, float* __restrict__ Acum, int lp, int nc)
{
  __shared__ unsigned short Bs[64][128];
  __shared__ unsigned short Cs[64][128];
  __shared__ unsigned short Xs[2][64][64];
  __shared__ unsigned short Ms[64][64];
  __shared__ float cum[2][64];
  __shared__ float dec[2][64];
  const int tid=threadIdx.x, lane=tid&31, wv=tid>>5;
  const int ch = blockIdx.x, b = blockIdx.y;
  const size_t row0 = (size_t)b*lp + ch*64;
  // async raw-copy B and C chunk tiles (16B per lane), VALU path for X*dt
  for (int e8 = tid; e8 < 1024; e8 += 256){
    int ll = e8 >> 4, n8 = (e8 & 15)*8;
    async_cp16(&Bs[ll][n8], &xconv[(row0+ll)*384 + 128 + n8]);
    async_cp16(&Cs[ll][n8], &xconv[(row0+ll)*384 + 256 + n8]);
  }
  for (int e = tid; e < 64*128; e += 256) {
    int ll = e >> 7, n = e & 127;
    int h = n >> 6, p = n & 63;
    Xs[h][ll][p] = f2b(b2f(xconv[(row0+ll)*384 + h*64 + p]) * dtsp[(row0+ll)*2 + h]);
  }
  async_wait();
  __syncthreads();
  if (tid < 2) {
    float s = 0.f;
    for (int ll=0; ll<64; ++ll){ s += dtA[(row0+ll)*2 + tid]; cum[tid][ll]=s; }
    Asum[((size_t)b*nc + ch)*2 + tid] = s;
  }
  __syncthreads();
  if (tid < 128) {
    int h = tid >> 6, ll = tid & 63;
    float cl = cum[h][ll];
    dec[h][ll] = expf(cum[h][63] - cl);
    Acum[(((size_t)b*nc + ch)*2 + h)*64 + ll] = cl;
  }
  __syncthreads();
  const int half = lane>>4, l16 = lane&15;
  const int tm = wv & 3, tnb = (wv>>2)*2;
  // G = C * B^T  (64x64, K=128)
  v8f g0={}, g1={};
  for (int ks=0; ks<4; ++ks){
    int k0 = ks*32;
    FragU a,b0,b1;
    int ar = tm*16 + l16;
    for (int i=0;i<8;++i){ a.u[i]=Cs[ar][k0+half*16+i]; a.u[8+i]=Cs[ar][k0+half*16+8+i]; }
    int c0 = tnb*16+l16, c1=c0+16;
    for (int i=0;i<16;++i){ b0.u[i]=Bs[c0][k0+half*16+i]; b1.u[i]=Bs[c1][k0+half*16+i]; }
    g0 = wmma_bf(a,b0,g0); g1 = wmma_bf(a,b1,g1);
  }
  for (int h=0; h<2; ++h) {
    for (int t=0;t<2;++t){
      v8f g = t? g1 : g0;
      int ss = (tnb+t)*16 + l16;
      for (int j=0;j<8;++j){
        int ll = tm*16 + j + 8*half;
        float v = (ll >= ss) ? g[j] * expf(cum[h][ll] - cum[h][ss]) : 0.f;
        Ms[ll][ss] = f2b(v);
      }
    }
    __syncthreads();
    // Y_diag = Ms * X_h (64x64, K=64)
    v8f y0={}, y1={};
    for (int ks=0;ks<2;++ks){
      int k0=ks*32;
      FragU a,b0,b1;
      int ar = tm*16+l16;
      for (int i=0;i<8;++i){ a.u[i]=Ms[ar][k0+half*16+i]; a.u[8+i]=Ms[ar][k0+half*16+8+i]; }
      int c0=tnb*16+l16, c1=c0+16;
      for (int i=0;i<16;++i){ b0.u[i]=Xs[h][k0+half*16+i][c0]; b1.u[i]=Xs[h][k0+half*16+i][c1]; }
      y0=wmma_bf(a,b0,y0); y1=wmma_bf(a,b1,y1);
    }
    float Dv = Dh[h];
    for (int t=0;t<2;++t){
      v8f y = t? y1:y0;
      int p = (tnb+t)*16 + l16;
      for (int j=0;j<8;++j){
        int ll = tm*16 + j + 8*half;
        size_t r = row0 + ll;
        float xr = b2f(xconv[r*384 + h*64 + p]);
        Y[r*128 + h*64 + p] = y[j] + xr*Dv;
      }
    }
    __syncthreads();
  }
  // states_h = X_h^T * (dec .* B) : M=64(p), N=128(n), K=64(l)
  for (int h=0; h<2; ++h) {
    const int stm = wv & 3;
    const int stnb = (wv>>2)*4;
    v8f s0={},s1={},s2={},s3={};
    for (int ks=0;ks<2;++ks){
      int k0=ks*32;
      FragU a, bb0, bb1, bb2, bb3;
      int pr = stm*16 + l16;
      for (int i=0;i<8;++i){ a.u[i]=Xs[h][k0+half*16+i][pr]; a.u[8+i]=Xs[h][k0+half*16+8+i][pr]; }
      FragU* bbs[4] = {&bb0,&bb1,&bb2,&bb3};
      for (int t=0;t<4;++t){
        int cn = (stnb+t)*16 + l16;
        for (int i=0;i<16;++i){
          int kk = k0 + half*16 + i;
          bbs[t]->u[i] = f2b(b2f(Bs[kk][cn]) * dec[h][kk]);
        }
      }
      s0=wmma_bf(a,bb0,s0); s1=wmma_bf(a,bb1,s1); s2=wmma_bf(a,bb2,s2); s3=wmma_bf(a,bb3,s3);
    }
    float* Sout = &Sbuf[(((size_t)b*nc + ch)*2 + h)*64*128];
    for (int t=0;t<4;++t){
      v8f sv = (t==0)?s0 : (t==1)?s1 : (t==2)?s2 : s3;
      int n = (stnb+t)*16 + l16;
      for (int j=0;j<8;++j){
        int p = stm*16 + j + 8*half;
        Sout[p*128 + n] = sv[j];
      }
    }
  }
}

// ---------------- inter-chunk state scan ----------------
__global__ __launch_bounds__(128) void k_ssd_scan(
    const float* __restrict__ Sbuf, const float* __restrict__ Asum,
    float* __restrict__ Sprev, int nc)
{
  int b = blockIdx.x >> 1, h = blockIdx.x & 1;
  int t = threadIdx.x;
  float S[64];
  for (int i=0;i<64;++i) S[i]=0.f;
  for (int z=0; z<nc; ++z){
    size_t base = (((size_t)b*nc + z)*2 + h)*8192;
    float a = expf(Asum[((size_t)b*nc+z)*2 + h]);
    for (int i=0;i<64;++i){
      Sprev[base + t*64 + i] = S[i];
      S[i] = a*S[i] + Sbuf[base + t*64 + i];
    }
  }
}

// ---------------- Y_off = diag(exp(Acum)) * C * Sprev^T ----------------
__global__ __launch_bounds__(256) void k_ssd_yoff(
    const unsigned short* __restrict__ xconv, const float* __restrict__ Sprev,
    const float* __restrict__ Acum, float* __restrict__ Y, int lp, int nc)
{
  __shared__ unsigned short Cs[64][128];
  __shared__ unsigned short Ss[64][128];
  const int tid=threadIdx.x, lane=tid&31, wv=tid>>5;
  const int ch=blockIdx.x, b=blockIdx.y;
  const size_t row0 = (size_t)b*lp + ch*64;
  for (int e8 = tid; e8 < 1024; e8 += 256){
    int ll = e8 >> 4, n8 = (e8 & 15)*8;
    async_cp16(&Cs[ll][n8], &xconv[(row0+ll)*384 + 256 + n8]);
  }
  async_wait();
  const int half=lane>>4, l16=lane&15, tm=wv&3, tnb=(wv>>2)*2;
  for (int h=0; h<2; ++h){
    __syncthreads();
    const float* Spp = &Sprev[(((size_t)b*nc+ch)*2+h)*8192];
    for (int e=tid; e<8192; e+=256) Ss[e>>7][e&127] = f2b(Spp[e]);
    __syncthreads();
    v8f y0={},y1={};
    for (int ks=0;ks<4;++ks){
      int k0=ks*32;
      FragU a,b0,b1;
      int ar=tm*16+l16;
      for(int i=0;i<8;++i){ a.u[i]=Cs[ar][k0+half*16+i]; a.u[8+i]=Cs[ar][k0+half*16+8+i]; }
      int c0=tnb*16+l16, c1=c0+16;
      for(int i=0;i<16;++i){ b0.u[i]=Ss[c0][k0+half*16+i]; b1.u[i]=Ss[c1][k0+half*16+i]; }
      y0=wmma_bf(a,b0,y0); y1=wmma_bf(a,b1,y1);
    }
    const float* ac = &Acum[(((size_t)b*nc+ch)*2+h)*64];
    for (int t=0;t<2;++t){
      v8f y=t?y1:y0;
      int p=(tnb+t)*16+l16;
      for(int j=0;j<8;++j){
        int ll=tm*16+j+8*half;
        size_t r=row0+ll;
        Y[r*128 + h*64 + p] += expf(ac[ll]) * y[j];
      }
    }
  }
}

// ---------------- gated RMSNorm ----------------
__global__ __launch_bounds__(128) void k_gate_norm(
    const float* __restrict__ Y, const unsigned short* __restrict__ proj,
    const float* __restrict__ norm_w, unsigned short* __restrict__ Yn, int ldp)
{
  __shared__ float red[4];
  size_t r = blockIdx.x;
  int c = threadIdx.x;
  float z = b2f(proj[r*ldp + c]);
  float g = z / (1.f + expf(-z));
  float y = Y[r*128 + c] * g;
  float s = y*y;
  for (int o=16;o>0;o>>=1) s += __shfl_down(s, o, 32);
  if ((c&31)==0) red[c>>5]=s;
  __syncthreads();
  float tot = red[0]+red[1]+red[2]+red[3];
  float rms = rsqrtf(tot/128.f + 1e-5f);
  Yn[r*128+c] = f2b(y * rms * norm_w[c]);
}

// ---------------- bilinear upsample into channels-last cat buffer ----------------
__global__ void k_upcat(const unsigned short* __restrict__ src, unsigned short* __restrict__ cat,
                        int ssrc, int lp, int chOff, int total)
{
  int i = blockIdx.x*256 + threadIdx.x; if (i >= total) return;
  int c = i & 127; int rest = i >> 7;
  int x = rest % 56; int y = (rest/56) % 56; int b = rest/3136;
  float sc = (float)ssrc / 56.f;
  float fy = (y+0.5f)*sc - 0.5f, fx = (x+0.5f)*sc - 0.5f;
  int y0 = (int)floorf(fy); float wy = fy - (float)y0;
  int x0 = (int)floorf(fx); float wx = fx - (float)x0;
  int y0c = min(max(y0,0), ssrc-1), y1c = min(max(y0+1,0), ssrc-1);
  int x0c = min(max(x0,0), ssrc-1), x1c = min(max(x0+1,0), ssrc-1);
  size_t rb = (size_t)b*lp;
  float v00 = b2f(src[(rb + y0c*ssrc + x0c)*128 + c]);
  float v01 = b2f(src[(rb + y0c*ssrc + x1c)*128 + c]);
  float v10 = b2f(src[(rb + y1c*ssrc + x0c)*128 + c]);
  float v11 = b2f(src[(rb + y1c*ssrc + x1c)*128 + c]);
  float v = (1.f-wy)*((1.f-wx)*v00 + wx*v01) + wy*((1.f-wx)*v10 + wx*v11);
  cat[((size_t)b*3136 + y*56 + x)*512 + chOff + c] = f2b(v);
}

// ---------------- decode (2-class 1x1) ----------------
__global__ void k_decode(const unsigned short* __restrict__ cat,
                         const float* __restrict__ dwp, const float* __restrict__ dbp,
                         float* __restrict__ dec, int total)
{
  int i = blockIdx.x*256 + threadIdx.x; if (i >= total) return;
  const unsigned short* r = &cat[(size_t)i*128];
  float a0 = dbp[0], a1 = dbp[1];
  for (int c=0;c<128;++c){ float v=b2f(r[c]); a0 += v*dwp[c]; a1 += v*dwp[128+c]; }
  dec[(size_t)i*2]   = a0;
  dec[(size_t)i*2+1] = a1;
}

// ---------------- final 4x bilinear upsample to d_out ----------------
__global__ void k_upfinal(const float* __restrict__ dec, float* __restrict__ out, int total)
{
  int i = blockIdx.x*256 + threadIdx.x; if (i >= total) return;
  int ox = i % 224; int oy = (i/224) % 224; int o = (i/(224*224)) % 2; int b = i/(2*224*224);
  float fy = (oy+0.5f)*0.25f - 0.5f, fx = (ox+0.5f)*0.25f - 0.5f;
  int y0 = (int)floorf(fy); float wy = fy - (float)y0;
  int x0 = (int)floorf(fx); float wx = fx - (float)x0;
  int y0c = min(max(y0,0),55), y1c = min(max(y0+1,0),55);
  int x0c = min(max(x0,0),55), x1c = min(max(x0+1,0),55);
  size_t rb = (size_t)b*3136;
  float v00 = dec[(rb + y0c*56 + x0c)*2 + o];
  float v01 = dec[(rb + y0c*56 + x1c)*2 + o];
  float v10 = dec[(rb + y1c*56 + x0c)*2 + o];
  float v11 = dec[(rb + y1c*56 + x1c)*2 + o];
  out[i] = (1.f-wy)*((1.f-wx)*v00 + wx*v01) + wy*((1.f-wx)*v10 + wx*v11);
}

// ================= host orchestration =================
extern "C" void kernel_launch(void* const* d_in, const int* in_sizes, int n_in,
                              void* d_out, int out_size, void* d_ws, size_t ws_size,
                              hipStream_t stream) {
  (void)in_sizes; (void)n_in; (void)out_size; (void)ws_size;
  int idx = 0;
  auto nxt = [&](){ return (const float*)d_in[idx++]; };
  const float *Ai[4], *Bi[4];
  for (int i=0;i<4;++i){ Ai[i]=nxt(); Bi[i]=nxt(); }
  struct MP { const float *in_proj,*conv_w,*conv_b,*dt_bias,*A_log,*Dv,*norm_w,*out_proj; };
  struct SP { const float *tw,*tg,*tb,*fiw,*fib; MP mf,mb; const float *fow,*fob; };
  SP sp[4];
  for (int i=0;i<4;++i){
    sp[i].tw=nxt(); sp[i].tg=nxt(); sp[i].tb=nxt();
    sp[i].fiw=nxt(); sp[i].fib=nxt();
    MP* ms[2]={&sp[i].mf,&sp[i].mb};
    for (int d=0;d<2;++d){ MP&m=*ms[d];
      m.in_proj=nxt(); m.conv_w=nxt(); m.conv_b=nxt(); m.dt_bias=nxt();
      m.A_log=nxt(); m.Dv=nxt(); m.norm_w=nxt(); m.out_proj=nxt(); }
    sp[i].fow=nxt(); sp[i].fob=nxt();
  }
  const float* ccw=nxt(); const float* ccg=nxt(); const float* ccb=nxt();
  const float* dwp=nxt(); const float* dbp=nxt();

  // workspace bump allocator (sized for scale 1: 50176 rows, nc=49)
  char* ws = (char*)d_ws; size_t off = 0;
  auto alloc = [&](size_t bytes)->char*{ char* p = ws + off; off = (off + bytes + 255) & ~(size_t)255; return p; };
  const size_t RMAX = 50176, NCMAX = 49;
  unsigned short* catbuf = (unsigned short*)alloc(RMAX*512*2);
  unsigned short* F      = (unsigned short*)alloc(RMAX*128*2);
  unsigned short* U      = (unsigned short*)alloc(RMAX*64*2);
  unsigned short* Ur     = (unsigned short*)alloc(RMAX*64*2);
  unsigned short* Pj     = (unsigned short*)alloc(RMAX*528*2);
  unsigned short* Xc     = (unsigned short*)alloc(RMAX*384*2);
  float* DTs  = (float*)alloc(RMAX*2*4);
  float* DTa  = (float*)alloc(RMAX*2*4);
  float* ACs  = (float*)alloc(16*NCMAX*2*64*4);
  float* ASm  = (float*)alloc(16*NCMAX*2*4);
  float* Sb   = (float*)alloc(16*NCMAX*2*8192*4);
  float* Spv  = (float*)alloc(16*NCMAX*2*8192*4);
  float* Yf   = (float*)alloc(RMAX*128*4);
  unsigned short* Yn = (unsigned short*)alloc(RMAX*128*2);
  unsigned short* Yc = (unsigned short*)alloc(RMAX*128*2);
  unsigned short* Yt = (unsigned short*)alloc(RMAX*64*2);
  unsigned short* Mo = (unsigned short*)alloc((size_t)16*832*128*2);
  unsigned short* Co = (unsigned short*)alloc(RMAX*128*2);
  unsigned short* Wc = (unsigned short*)alloc((size_t)128*9216*2);
  unsigned short* Wsm= (unsigned short*)alloc((size_t)640*64*2);  // padded for async OOB tile reads
  float* Dec = (float*)alloc(RMAX*2*4);

  const int cL[4]={64,128,256,512}, sL[4]={56,28,14,7}, lpL[4]={3136,832,256,64}, ncL[4]={49,13,4,1};

  for (int i=0;i<4;++i){
    const int c=cL[i], s=sL[i], lp=lpL[i], nc=ncL[i];
    const int l=s*s, R=16*lp, Kc=2*c*9;
    // conv weights -> bf16 ; zero F (pad rows must be zero)
    k_cvt<<<CDIV(128*Kc,256),256,0,stream>>>(sp[i].tw, Wc, 128*Kc);
    k_zero16<<<CDIV((long long)R*128,256),256,0,stream>>>(F, (long long)R*128);
    k_conv_nchw2<<<dim3(1, CDIV(16*l,64)),256,0,stream>>>(Ai[i], Bi[i], Wc, F, sp[i].tg, sp[i].tb, c, s, lp);
    // u = F @ fc_in_w^T + b
    k_cvt<<<CDIV(64*128,256),256,0,stream>>>(sp[i].fiw, Wsm, 64*128);
    k_gemm<3,64><<<dim3(1, R/64),256,0,stream>>>(F, Wsm, U, R,64,128, 128,128,64, sp[i].fib, nullptr);
    k_rev<<<CDIV(R*64,256),256,0,stream>>>(U, Ur, lp, R*64);
    for (int d=0; d<2; ++d){
      const MP& m = d ? sp[i].mb : sp[i].mf;
      const unsigned short* Ud = d ? Ur : U;
      k_cvt<<<CDIV(514*64,256),256,0,stream>>>(m.in_proj, Wsm, 514*64);
      k_gemm<1,128><<<dim3(5, R/64),256,0,stream>>>(Ud, Wsm, Pj, R,514,64, 64,64,528, nullptr,nullptr);
      k_dwconv<<<CDIV(R*384,256),256,0,stream>>>(Pj, m.conv_w, m.conv_b, Xc, lp, R*384);
      k_dt<<<CDIV(R*2,256),256,0,stream>>>(Pj, m.dt_bias, m.A_log, DTs, DTa, R*2);
      k_ssd_chunk<<<dim3(nc,16),256,0,stream>>>(Xc, DTs, DTa, m.Dv, Yf, Sb, ASm, ACs, lp, nc);
      k_ssd_scan<<<32,128,0,stream>>>(Sb, ASm, Spv, nc);
      k_ssd_yoff<<<dim3(nc,16),256,0,stream>>>(Xc, Spv, ACs, Yf, lp, nc);
      k_gate_norm<<<R,128,0,stream>>>(Yf, Pj, m.norm_w, Yn, 528);
      k_cvt<<<CDIV(64*128,256),256,0,stream>>>(m.out_proj, Wsm, 64*128);
      if (d==0)
        k_gemm<1,64><<<dim3(1, R/64),256,0,stream>>>(Yn, Wsm, Yc, R,64,128, 128,128,128, nullptr,nullptr);
      else {
        k_gemm<1,64><<<dim3(1, R/64),256,0,stream>>>(Yn, Wsm, Yt, R,64,128, 128,128,64, nullptr,nullptr);
        k_revcat<<<CDIV(R*64,256),256,0,stream>>>(Yt, Yc, lp, R*64);
      }
    }
    // fc_out
    k_cvt<<<CDIV(128*128,256),256,0,stream>>>(sp[i].fow, Wsm, 128*128);
    if (i==0)
      k_gemm<3,128><<<dim3(1, R/64),256,0,stream>>>(Yc, Wsm, catbuf, R,128,128, 128,128,512, sp[i].fob, nullptr);
    else {
      k_gemm<3,128><<<dim3(1, R/64),256,0,stream>>>(Yc, Wsm, Mo, R,128,128, 128,128,128, sp[i].fob, nullptr);
      k_upcat<<<CDIV(16*3136*128,256),256,0,stream>>>(Mo, catbuf, s, lp, i*128, 16*3136*128);
    }
  }
  // catconv + decode + final upsample
  k_cvt<<<CDIV(128*4608,256),256,0,stream>>>(ccw, Wc, 128*4608);
  k_conv_cl<<<dim3(1, CDIV(16*3136,64)),256,0,stream>>>(catbuf, Wc, Co, ccg, ccb, 512, 56);
  k_decode<<<CDIV(50176,256),256,0,stream>>>(Co, dwp, dbp, Dec, 50176);
  k_upfinal<<<CDIV(16*2*224*224,256),256,0,stream>>>(Dec, (float*)d_out, 16*2*224*224);
}
